// MPNNet_v2_31233002176779
// MI455X (gfx1250) — matrix operand order repaired and checked by reading, not compiled, not run on hardware
//
#include <hip/hip_runtime.h>
#include <math.h>

// ---------------- problem constants ----------------
#define NN      32768      // nodes
#define NE      65536      // edges (power of two)
#define NB      512        // graphs (64 nodes each, contiguous)
#define DIMF    64
#define NFEAT   14
#define EDIM    4
#define NHID    128
#define CSTEPS  12
#define S2S     3
#define NSTEM   1024
#define NOSTEM  105

typedef unsigned short u16;
typedef __attribute__((ext_vector_type(16))) __bf16 v16bf;
typedef __attribute__((ext_vector_type(8)))  float  v8f;

union AFrag { v16bf v; uint4 q[2]; u16 s[16]; };

__device__ __forceinline__ u16 f2bf(float f) {
  union { float f; unsigned u; } v; v.f = f;
  unsigned u = v.u;
  return (u16)((u + 0x7FFFu + ((u >> 16) & 1u)) >> 16);   // RNE
}
__device__ __forceinline__ float bf2f(u16 b) {
  union { unsigned u; float f; } v; v.u = ((unsigned)b) << 16; return v.f;
}
__device__ __forceinline__ float lrelu(float x) { return x > 0.f ? x : 0.01f * x; }
__device__ __forceinline__ float sigf(float x)  { return 1.f / (1.f + expf(-x)); }

// ---------------- trivial elementwise kernels ----------------
__global__ void k_zero(float* p, int n) {
  int i = blockIdx.x * blockDim.x + threadIdx.x;
  if (i < n) p[i] = 0.f;
}

__global__ void k_deg(const int* dst, float* deg) {
  int e = blockIdx.x * blockDim.x + threadIdx.x;
  if (e < NE) atomicAdd(&deg[dst[e]], 1.0f);
}

__global__ void k_invdeg(const float* deg, float* inv) {
  int i = blockIdx.x * blockDim.x + threadIdx.x;
  if (i < NN) inv[i] = 1.0f / fmaxf(deg[i], 1.0f);
}

// lin0: X = leaky(x @ lin0_w.T + b)   [N,14]->[N,64]
__global__ void k_lin0(const float* x, const float* w, const float* b, float* X) {
  int t = blockIdx.x * blockDim.x + threadIdx.x;       // N*64 threads
  int node = t >> 6, d = t & 63;
  float a = b[d];
  const float* xr = x + node * NFEAT;
  const float* wr = w + d * NFEAT;
#pragma unroll
  for (int k = 0; k < NFEAT; k++) a = fmaf(xr[k], wr[k], a);
  X[t] = lrelu(a);
}

// edge hidden, TRANSPOSED: HEBT[c][e] = bf16(leaky(edge_attr @ net_w1.T + b1))
__global__ void k_ehid(const float* ea, const float* w1, const float* b1, u16* HEBT) {
  int t = blockIdx.x * blockDim.x + threadIdx.x;       // t = c*NE + e
  int c = t >> 16;                                     // NE == 1<<16
  int e = t & (NE - 1);
  float a = b1[c];
  const float* er = ea + e * EDIM;
  const float* wr = w1 + c * EDIM;
#pragma unroll
  for (int k = 0; k < EDIM; k++) a = fmaf(er[k], wr[k], a);
  HEBT[t] = f2bf(lrelu(a));
}

// Pack a weight matrix into WMMA *B*-fragment order (lane = N, split-K runs):
//   B[k][n] = trans ? src[n][k] : src[k][n]
//   dst[((nt*(K/32)+kc)*32 + lane)*16 + j] = B[kc*32 + kidxA(lane,j)][nt*16 + lane%16]
__global__ void k_packB(const float* src, u16* dst, int K, int Nout, int trans) {
  int tid = blockIdx.x * blockDim.x + threadIdx.x;
  int kcn = K >> 5;
  int total = (Nout >> 4) * kcn * 512;
  if (tid >= total) return;
  int j  = tid & 15;
  int l  = (tid >> 4) & 31;
  int kc = (tid >> 9) % kcn;
  int nt = tid / (kcn * 512);
  int k = kc * 32 + ((l >> 4) * 8) + (j & 7) + ((j >> 3) * 16), n = nt * 16 + (l & 15);
  float v = trans ? src[n * K + k] : src[k * Nout + n];
  dst[tid] = f2bf(v);
}

// Pack src[M,K] (row-major) into WMMA *A*-fragment order (lane = M, split-K runs):
//   dst[((mt*(K/32)+kc)*32 + lane)*16 + j] = src[mt*16 + lane%16][kc*32 + kidxA(lane,j)]
__global__ void k_packA(const float* src, u16* dst, int K, int Mrows) {
  int tid = blockIdx.x * blockDim.x + threadIdx.x;
  int kcn = K >> 5;
  int total = (Mrows >> 4) * kcn * 512;
  if (tid >= total) return;
  int j  = tid & 15;
  int l  = (tid >> 4) & 31;
  int kc = (tid >> 9) % kcn;
  int mt = tid / (kcn * 512);
  int m = mt * 16 + (l & 15);
  int kk = kc * 32 + ((l >> 4) * 8) + (j & 7) + ((j >> 3) * 16);
  dst[tid] = f2bf(src[(size_t)m * K + kk]);
}

// ---------------- W GEMM (transposed orientation -> packed b128 stores) -------------
// D[m=col, n=edge]: A = net_w2 A-fragments (per-block invariant),
//                   B = HEBT fragments (2 x b128 each). Lane stores 8 consecutive
//                   columns of one edge row of W -> single global_store_b128.
__global__ __launch_bounds__(32) void k_wgemm(const u16* __restrict__ HEBT,
                                              const u16* __restrict__ W2AF,
                                              const float* __restrict__ b2,
                                              u16* __restrict__ WBF) {
  int l = threadIdx.x, hi = l >> 4, c16 = l & 15;
  int ntg = blockIdx.x;     // 0..127: column-tile pair (nt = ntg*2 + t2)
  int etg = blockIdx.y;     // 0..127: group of 32 edge tiles

  v16bf Af[2][4];
  float bias[2][8];
#pragma unroll
  for (int t2 = 0; t2 < 2; t2++) {
    int nt = ntg * 2 + t2;
#pragma unroll
    for (int kc = 0; kc < 4; kc++)
      Af[t2][kc] = *(const v16bf*)(W2AF + ((size_t)(nt * 4 + kc) * 32 + l) * 16);
    const float* bp = b2 + nt * 16 + 8 * hi;
#pragma unroll
    for (int v = 0; v < 8; v++) bias[t2][v] = bp[v];
  }

  for (int t = 0; t < 32; t++) {
    int et = etg * 32 + t;
    AFrag B[4];
#pragma unroll
    for (int kc = 0; kc < 4; kc++) {
#pragma unroll
      for (int j = 0; j < 16; j++)
        B[kc].s[j] = HEBT[(size_t)(kc * 32 + 8 * hi + (j & 7) + 16 * (j >> 3)) * NE + et * 16 + c16];
    }
#pragma unroll
    for (int t2 = 0; t2 < 2; t2++) {
      v8f c = {};
#pragma unroll
      for (int kc = 0; kc < 4; kc++)
        c = __builtin_amdgcn_wmma_f32_16x16x32_bf16(false, Af[t2][kc], false,
                                                    B[kc].v, (short)0, c,
                                                    false, false);
      uint4 o;
      unsigned* op = (unsigned*)&o;
#pragma unroll
      for (int p2 = 0; p2 < 4; p2++)
        op[p2] = (unsigned)f2bf(c[2 * p2] + bias[t2][2 * p2]) |
                 ((unsigned)f2bf(c[2 * p2 + 1] + bias[t2][2 * p2 + 1]) << 16);
      // lane owns edge et*16+c16, columns (ntg*2+t2)*16 + 8*hi .. +7
      u16* dp = WBF + (size_t)(et * 16 + c16) * 4096 + (ntg * 2 + t2) * 16 + 8 * hi;
      *(uint4*)dp = o;
    }
  }
}

// ---------------- per-edge message + scatter-mean numerator ----------------
// wave-per-edge: 16 x b128 coalesced loads stream W_e; shfl-xor cross-group
// reduction; 2 fully-occupied fp32 atomics per lane into L2-resident AGG.
__global__ __launch_bounds__(256) void k_msg(const float* __restrict__ X,
                                             const u16* __restrict__ WBF,
                                             const int* __restrict__ src,
                                             const int* __restrict__ dst,
                                             float* __restrict__ AGG) {
  __shared__ float s[8][64];
  int l = threadIdx.x & 31, w = threadIdx.x >> 5;
  int e = blockIdx.x * 8 + w;
  int sn = src[e];
  s[w][l]      = X[(size_t)sn * DIMF + l];
  s[w][l + 32] = X[(size_t)sn * DIMF + l + 32];
  __syncthreads();

  int rowoff = l >> 3;          // 0..3 : which of 4 rows this lane covers
  int colg   = (l & 7) * 8;     // 8-column group
  const u16* wp = WBF + (size_t)e * 4096 + colg;
  float acc[8] = {0.f, 0.f, 0.f, 0.f, 0.f, 0.f, 0.f, 0.f};
#pragma unroll
  for (int t = 0; t < 16; t++) {
    int row = t * 4 + rowoff;
    uint4 q = *(const uint4*)(wp + (size_t)row * 64);   // 8 bf16, wave = 512B
    float si = s[w][row];
    const u16* pv = (const u16*)&q;
#pragma unroll
    for (int j = 0; j < 8; j++) acc[j] = fmaf(si, bf2f(pv[j]), acc[j]);
  }
  // reduce the 4 row-phases (lanes differing in bits 3 and 4)
#pragma unroll
  for (int j = 0; j < 8; j++) {
    acc[j] += __shfl_xor(acc[j], 8, 32);
    acc[j] += __shfl_xor(acc[j], 16, 32);
  }
  int dn = dst[e];
  int jj = 2 * rowoff;          // each row-phase group commits 2 of its 8 columns
  atomicAdd(&AGG[(size_t)dn * DIMF + colg + jj],     acc[jj]);
  atomicAdd(&AGG[(size_t)dn * DIMF + colg + jj + 1], acc[jj + 1]);
}

// ---------------- node update: NNConv root + leaky + GRU, all matmuls via WMMA ------
// one wave per 16-node tile; X/AGG staged through LDS with coalesced float4 I/O
__global__ __launch_bounds__(32) void k_update(float* __restrict__ X,
                                               const float* __restrict__ AGG,
                                               const float* __restrict__ INV,
                                               const u16* __restrict__ ROOTF,
                                               const u16* __restrict__ WIHF,
                                               const u16* __restrict__ WHHF,
                                               const float* __restrict__ cbias,
                                               const float* __restrict__ bih,
                                               const float* __restrict__ bhh) {
  __shared__ float xt[16 * 68];
  __shared__ float at[16 * 68];
  __shared__ float mt[16 * 68];
  __shared__ float sinv[16];
  int l = threadIdx.x, hi = l >> 4, c16 = l & 15;
  int tb = blockIdx.x * 16;

  // cooperative coalesced tile loads
#pragma unroll
  for (int ch = 0; ch < 8; ch++) {
    int flat = ch * 128 + l * 4;
    int row = flat >> 6, col = flat & 63;
    *(float4*)&xt[row * 68 + col] = *(const float4*)&X[(size_t)(tb + row) * DIMF + col];
    *(float4*)&at[row * 68 + col] = *(const float4*)&AGG[(size_t)(tb + row) * DIMF + col];
  }
  if (l < 16) sinv[l] = INV[tb + l];
  __syncthreads();

  // A-fragments of h (=X tile), K=64 -> 2 chunks of 32
  v16bf hA[2];
#pragma unroll
  for (int kc = 0; kc < 2; kc++) {
    AFrag a;
    const float* p = &xt[c16 * 68 + kc * 32 + (hi ? 8 : 0)];
#pragma unroll
    for (int j = 0; j < 8; j++) a.s[j] = f2bf(p[j]);
#pragma unroll
    for (int j = 0; j < 8; j++) a.s[8 + j] = f2bf(p[16 + j]);
    hA[kc] = a.v;
  }

  // root term: X @ conv_root  (4 n-tiles x 2 k-steps)
  v8f acc[4];
#pragma unroll
  for (int nt = 0; nt < 4; nt++) {
    v8f c = {};
#pragma unroll
    for (int kc = 0; kc < 2; kc++) {
      v16bf b = *(const v16bf*)(ROOTF + ((size_t)(nt * 2 + kc) * 32 + l) * 16);
      c = __builtin_amdgcn_wmma_f32_16x16x32_bf16(false, hA[kc], false, b,
                                                  (short)0, c, false, false);
    }
    acc[nt] = c;
  }

  // m = leaky(agg*inv_deg + root + bias) -> LDS (C-layout scatter, conflict-free)
#pragma unroll
  for (int nt = 0; nt < 4; nt++) {
    int col = nt * 16 + c16;
    float cb = cbias[col];
#pragma unroll
    for (int v = 0; v < 8; v++) {
      int row = v + 8 * hi;
      float a = at[row * 68 + col] * sinv[row];
      mt[row * 68 + col] = lrelu(a + acc[nt][v] + cb);
    }
  }
  __syncthreads();

  // A-fragments of m from LDS
  v16bf mA[2];
#pragma unroll
  for (int kc = 0; kc < 2; kc++) {
    AFrag a;
    const float* p = &mt[c16 * 68 + kc * 32 + (hi ? 8 : 0)];
#pragma unroll
    for (int j = 0; j < 8; j++) a.s[j] = f2bf(p[j]);
#pragma unroll
    for (int j = 0; j < 8; j++) a.s[8 + j] = f2bf(p[16 + j]);
    mA[kc] = a.v;
  }

  // GRU gates, 4 column-groups of 16; gate rows: r=g, z=g+4, n=g+8 (of 12 n-tiles)
#pragma unroll
  for (int g = 0; g < 4; g++) {
    v8f cir = {}, ciz = {}, cin = {}, chr_ = {}, chz = {}, chn = {};
#pragma unroll
    for (int kc = 0; kc < 2; kc++) {
      v16bf bir = *(const v16bf*)(WIHF + ((size_t)((g)     * 2 + kc) * 32 + l) * 16);
      v16bf biz = *(const v16bf*)(WIHF + ((size_t)((g + 4) * 2 + kc) * 32 + l) * 16);
      v16bf bin = *(const v16bf*)(WIHF + ((size_t)((g + 8) * 2 + kc) * 32 + l) * 16);
      v16bf bhr = *(const v16bf*)(WHHF + ((size_t)((g)     * 2 + kc) * 32 + l) * 16);
      v16bf bhz = *(const v16bf*)(WHHF + ((size_t)((g + 4) * 2 + kc) * 32 + l) * 16);
      v16bf bhn = *(const v16bf*)(WHHF + ((size_t)((g + 8) * 2 + kc) * 32 + l) * 16);
      cir  = __builtin_amdgcn_wmma_f32_16x16x32_bf16(false, mA[kc], false, bir, (short)0, cir,  false, false);
      ciz  = __builtin_amdgcn_wmma_f32_16x16x32_bf16(false, mA[kc], false, biz, (short)0, ciz,  false, false);
      cin  = __builtin_amdgcn_wmma_f32_16x16x32_bf16(false, mA[kc], false, bin, (short)0, cin,  false, false);
      chr_ = __builtin_amdgcn_wmma_f32_16x16x32_bf16(false, hA[kc], false, bhr, (short)0, chr_, false, false);
      chz  = __builtin_amdgcn_wmma_f32_16x16x32_bf16(false, hA[kc], false, bhz, (short)0, chz,  false, false);
      chn  = __builtin_amdgcn_wmma_f32_16x16x32_bf16(false, hA[kc], false, bhn, (short)0, chn,  false, false);
    }
    int dcol = g * 16 + c16;
    float b_ir = bih[dcol],       b_hr = bhh[dcol];
    float b_iz = bih[64 + dcol],  b_hz = bhh[64 + dcol];
    float b_in = bih[128 + dcol], b_hn = bhh[128 + dcol];
#pragma unroll
    for (int v = 0; v < 8; v++) {
      int row = v + 8 * hi;
      float hv = xt[row * 68 + dcol];
      float r  = sigf(cir[v] + b_ir + chr_[v] + b_hr);
      float z  = sigf(ciz[v] + b_iz + chz[v] + b_hz);
      float n  = tanhf(cin[v] + b_in + r * (chn[v] + b_hn));
      xt[row * 68 + dcol] = (1.f - z) * n + z * hv;   // overwrite with h_new
    }
  }
  __syncthreads();

  // cooperative coalesced store of h_new
#pragma unroll
  for (int ch = 0; ch < 8; ch++) {
    int flat = ch * 128 + l * 4;
    int row = flat >> 6, col = flat & 63;
    *(float4*)&X[(size_t)(tb + row) * DIMF + col] = *(const float4*)&xt[row * 68 + col];
  }
}

// ---------------- Set2Set (graphs are contiguous 64-node blocks) ----------------
__global__ __launch_bounds__(64) void k_s2s_lstm(const float* QSTAR, float* HH, float* CC,
                                                 const float* wih, const float* whh,
                                                 const float* bihv, const float* bhhv) {
  __shared__ float sh[64];
  int g = blockIdx.x, d = threadIdx.x;
  sh[d] = HH[g * 64 + d];
  __syncthreads();
  const float* q = QSTAR + g * 128;
  float gate[4];
#pragma unroll
  for (int gg = 0; gg < 4; gg++) {                 // torch order: i,f,g,o
    int row = gg * 64 + d;
    float a = bihv[row] + bhhv[row];
    const float* wi = wih + (size_t)row * 128;
    for (int k = 0; k < 128; k++) a = fmaf(q[k], wi[k], a);
    const float* wh = whh + (size_t)row * 64;
    for (int k = 0; k < 64; k++) a = fmaf(sh[k], wh[k], a);
    gate[gg] = a;
  }
  float c = CC[g * 64 + d];
  c = sigf(gate[1]) * c + sigf(gate[0]) * tanhf(gate[2]);
  CC[g * 64 + d] = c;
  HH[g * 64 + d] = sigf(gate[3]) * tanhf(c);
}

__global__ __launch_bounds__(64) void k_s2s_attend(const float* __restrict__ X,
                                                   const float* __restrict__ HH,
                                                   float* __restrict__ QSTAR) {
  __shared__ float red[64];
  __shared__ float sa[64];
  __shared__ float sq[64];
  int g = blockIdx.x, t = threadIdx.x;
  sq[t] = HH[g * 64 + t];
  __syncthreads();
  const float* xr = X + (size_t)(g * 64 + t) * DIMF;
  float e = 0.f;
  for (int k = 0; k < 64; k++) e = fmaf(xr[k], sq[k], e);
  red[t] = e; __syncthreads();
  for (int s = 32; s > 0; s >>= 1) { if (t < s) red[t] = fmaxf(red[t], red[t + s]); __syncthreads(); }
  float mx = red[0]; __syncthreads();
  float ex = expf(e - mx);
  red[t] = ex; __syncthreads();
  for (int s = 32; s > 0; s >>= 1) { if (t < s) red[t] += red[t + s]; __syncthreads(); }
  float sum = red[0]; __syncthreads();
  sa[t] = ex / sum; __syncthreads();
  float r = 0.f;
  for (int n = 0; n < 64; n++) r = fmaf(sa[n], X[(size_t)(g * 64 + n) * DIMF + t], r);
  QSTAR[g * 128 + t]      = sq[t];
  QSTAR[g * 128 + 64 + t] = r;
}

__global__ void k_sout(const float* QSTAR, const float* w, const float* b, float* out) {
  int g = blockIdx.x * blockDim.x + threadIdx.x;
  if (g >= NB) return;
  float a = b[0];
  for (int k = 0; k < 128; k++) a = fmaf(QSTAR[g * 128 + k], w[k], a);
  out[g] = a;
}

// ---------------- per-stem head ----------------
__global__ __launch_bounds__(128) void k_stem(const float* __restrict__ X, const int* sidx,
                                              const float* w1, const float* b1,
                                              const float* w2, const float* b2,
                                              float* out) {
  __shared__ float sf[64];
  __shared__ float hid[512];
  int s = blockIdx.x, t = threadIdx.x;
  int node = sidx[s];
  if (t < 64) sf[t] = X[(size_t)node * DIMF + t];
  __syncthreads();
  for (int r = t; r < 512; r += 128) {
    float a = b1[r];
    const float* wr = w1 + (size_t)r * 64;
    for (int k = 0; k < 64; k++) a = fmaf(sf[k], wr[k], a);
    hid[r] = lrelu(a);
  }
  __syncthreads();
  for (int r = t; r < NOSTEM; r += 128) {
    float a = b2[r];
    const float* wr = w2 + (size_t)r * 512;
    for (int k = 0; k < 512; k++) a = fmaf(hid[k], wr[k], a);
    out[(size_t)s * NOSTEM + r] = a;
  }
}

// ---------------- host orchestration ----------------
static inline size_t alup(size_t x) { return (x + 255) & ~(size_t)255; }

extern "C" void kernel_launch(void* const* d_in, const int* in_sizes, int n_in,
                              void* d_out, int out_size, void* d_ws, size_t ws_size,
                              hipStream_t stream) {
  const float* x       = (const float*)d_in[0];
  const float* eattr   = (const float*)d_in[1];
  const int*   eidx    = (const int*)  d_in[2];   // [2,E]: src then dst
  const int*   sidx    = (const int*)  d_in[4];
  const float* lin0_w  = (const float*)d_in[5];
  const float* lin0_b  = (const float*)d_in[6];
  const float* net_w1  = (const float*)d_in[7];
  const float* net_b1  = (const float*)d_in[8];
  const float* net_w2  = (const float*)d_in[9];
  const float* net_b2  = (const float*)d_in[10];
  const float* c_root  = (const float*)d_in[11];
  const float* c_bias  = (const float*)d_in[12];
  const float* gwih    = (const float*)d_in[13];
  const float* gwhh    = (const float*)d_in[14];
  const float* gbih    = (const float*)d_in[15];
  const float* gbhh    = (const float*)d_in[16];
  const float* l1w     = (const float*)d_in[17];
  const float* l1b     = (const float*)d_in[18];
  const float* l2w     = (const float*)d_in[19];
  const float* l2b     = (const float*)d_in[20];
  const float* swih    = (const float*)d_in[21];
  const float* swhh    = (const float*)d_in[22];
  const float* sbih    = (const float*)d_in[23];
  const float* sbhh    = (const float*)d_in[24];
  const float* l3w     = (const float*)d_in[25];
  const float* l3b     = (const float*)d_in[26];
  const int* esrc = eidx;
  const int* edst = eidx + NE;

  // workspace layout (~573 MB)
  char* base = (char*)d_ws;
  size_t off = 0;
  float* X    = (float*)(base + off); off = alup(off + (size_t)NN * DIMF * 4);
  float* AGG  = (float*)(base + off); off = alup(off + (size_t)NN * DIMF * 4);
  float* DEG  = (float*)(base + off); off = alup(off + (size_t)NN * 4);
  float* INV  = (float*)(base + off); off = alup(off + (size_t)NN * 4);
  u16*   HEBT = (u16*)  (base + off); off = alup(off + (size_t)NHID * NE * 2);
  u16*   W2AF = (u16*)  (base + off); off = alup(off + (size_t)256 * 4 * 512 * 2);
  u16*   ROOTF= (u16*)  (base + off); off = alup(off + (size_t)4 * 2 * 512 * 2);
  u16*   WIHF = (u16*)  (base + off); off = alup(off + (size_t)12 * 2 * 512 * 2);
  u16*   WHHF = (u16*)  (base + off); off = alup(off + (size_t)12 * 2 * 512 * 2);
  float* QSTAR= (float*)(base + off); off = alup(off + (size_t)NB * 128 * 4);
  float* HH   = (float*)(base + off); off = alup(off + (size_t)NB * 64 * 4);
  float* CC   = (float*)(base + off); off = alup(off + (size_t)NB * 64 * 4);
  u16*   WBF  = (u16*)  (base + off); off = alup(off + (size_t)NE * 4096 * 2);
  (void)ws_size; (void)in_sizes; (void)n_in; (void)out_size;

  float* out_s = (float*)d_out;            // [512] graph scores
  float* out_p = (float*)d_out + NB;       // [1024*105] per-stem

  // degree / inv_deg
  k_zero<<<(NN + 255) / 256, 256, 0, stream>>>(DEG, NN);
  k_deg<<<(NE + 255) / 256, 256, 0, stream>>>(edst, DEG);
  k_invdeg<<<(NN + 255) / 256, 256, 0, stream>>>(DEG, INV);

  // lin0 + edge hidden (transposed)
  k_lin0<<<(NN * DIMF) / 256, 256, 0, stream>>>(x, lin0_w, lin0_b, X);
  k_ehid<<<(NE * NHID) / 256, 256, 0, stream>>>(eattr, net_w1, net_b1, HEBT);

  // pack weight fragments (bf16)
  k_packA<<<(256 * 4 * 512 + 255) / 256, 256, 0, stream>>>(net_w2, W2AF, 128, 4096);
  k_packB<<<(4 * 2 * 512 + 255) / 256, 256, 0, stream>>>(c_root, ROOTF, 64, 64, 0);
  k_packB<<<(12 * 2 * 512 + 255) / 256, 256, 0, stream>>>(gwih, WIHF, 64, 192, 1);
  k_packB<<<(12 * 2 * 512 + 255) / 256, 256, 0, stream>>>(gwhh, WHHF, 64, 192, 1);

  // big WMMA GEMM: W (bf16, 512MB), hoisted out of the conv loop
  k_wgemm<<<dim3(128, 128), 32, 0, stream>>>(HEBT, W2AF, net_b2, WBF);

  // 12 conv steps: zero agg -> message/scatter -> WMMA node update (out==h invariant)
  for (int s = 0; s < CSTEPS; s++) {
    k_zero<<<(NN * DIMF) / 256, 256, 0, stream>>>(AGG, NN * DIMF);
    k_msg<<<NE / 8, 256, 0, stream>>>(X, WBF, esrc, edst, AGG);
    k_update<<<NN / 16, 32, 0, stream>>>(X, AGG, INV, ROOTF, WIHF, WHHF,
                                         c_bias, gbih, gbhh);
  }

  // Set2Set
  k_zero<<<(NB * 128 + 255) / 256, 256, 0, stream>>>(QSTAR, NB * 128);
  k_zero<<<(NB * 64 + 255) / 256, 256, 0, stream>>>(HH, NB * 64);
  k_zero<<<(NB * 64 + 255) / 256, 256, 0, stream>>>(CC, NB * 64);
  for (int t = 0; t < S2S; t++) {
    k_s2s_lstm<<<NB, 64, 0, stream>>>(QSTAR, HH, CC, swih, swhh, sbih, sbhh);
    k_s2s_attend<<<NB, 64, 0, stream>>>(X, HH, QSTAR);
  }
  k_sout<<<(NB + 255) / 256, 256, 0, stream>>>(QSTAR, l3w, l3b, out_s);

  // stem head
  k_stem<<<NSTEM, 128, 0, stream>>>(X, sidx, l1w, l1b, l2w, l2b, out_p);
}
